// MultiHeadAttention_46901042872697
// MI455X (gfx1250) — matrix operand order repaired and checked by reading, not compiled, not run on hardware
//
#include <hip/hip_runtime.h>
#include <hip/hip_bf16.h>

// ---------------- problem constants ----------------
constexpr int Bz = 4, T = 2048, C = 1024, H = 16, HS = 64;

typedef __bf16 v16bf __attribute__((ext_vector_type(16)));
typedef float  v8f   __attribute__((ext_vector_type(8)));
typedef unsigned int u32x4 __attribute__((ext_vector_type(4)));
typedef int          i32x8 __attribute__((ext_vector_type(8)));
typedef int          i32x4 __attribute__((ext_vector_type(4)));

#if defined(__has_builtin)
#if __has_builtin(__builtin_amdgcn_tensor_load_to_lds)
#define USE_TDM 1
#endif
#endif

// ---------------- helpers ----------------
__device__ inline v8f zero8() {
    v8f z; for (int i = 0; i < 8; ++i) z[i] = 0.0f; return z;
}

__device__ inline v16bf load_frag(const __bf16* p0, const __bf16* p1) {
    v16bf f;
    uint4* d = reinterpret_cast<uint4*>(&f);
    d[0] = *reinterpret_cast<const uint4*>(p0);
    d[1] = *reinterpret_cast<const uint4*>(p1);
    return f;
}

// A-matrix (16x32, row-major, stride in elements). ISA layout:
// lane l: row m = l&15 ; K = kBase..kBase+7 and kBase+16..kBase+23 (kBase = 8*(l>=16))
__device__ inline v16bf load_fragA(const __bf16* base, int stride, int lane) {
    int m  = lane & 15;
    int kb = (lane >> 4) << 3;
    const __bf16* p = base + m * stride + kb;
    return load_frag(p, p + 16);
}

// B-matrix (32x16) read from an N-major buffer Bt[n][k]. ISA layout:
// lane l: col n = l&15 ; K = kBase..kBase+7 and kBase+16..kBase+23 (kBase = 8*(l>=16))
__device__ inline v16bf load_fragB(const __bf16* base, int stride, int lane) {
    int n  = lane & 15;
    int kh = (lane >> 4) << 3;
    const __bf16* p = base + n * stride + kh;
    return load_frag(p, p + 16);
}

__device__ inline v8f wmma_bf16(v16bf a, v16bf b, v8f c) {
    return __builtin_amdgcn_wmma_f32_16x16x32_bf16(false, a, false, b,
                                                   (short)0, c, false, false);
}

__device__ inline float rowmax16(float v) {
    for (int m = 1; m < 16; m <<= 1) v = fmaxf(v, __shfl_xor(v, m, 32));
    return v;
}
__device__ inline float rowsum16(float v) {
    for (int m = 1; m < 16; m <<= 1) v += __shfl_xor(v, m, 32);
    return v;
}

#ifdef USE_TDM
// LDS byte offset of a __shared__ object: low 32 bits of the generic (shared
// aperture) address are the wave-relative LDS byte address (ISA 10.2).
__device__ inline unsigned lds_off(const void* p) {
    return (unsigned)(unsigned long long)p;
}

// TDM 2D tile load: d1 rows of d0 bf16 elements, row stride in elements.
// LDS padding: after 2^(padInterval+1) DWORDs stored insert (padAmount+1) DWORDs.
__device__ inline void tdm_load_2d(unsigned ldsByteOff, const void* gptr,
                                   unsigned d0, unsigned d1, unsigned strideElems,
                                   unsigned padInterval, unsigned padAmount) {
    unsigned long long ga = (unsigned long long)gptr;
    u32x4 g0;
    g0[0] = 1u;                                   // count=1 (valid descriptor)
    g0[1] = ldsByteOff;                           // lds_addr
    g0[2] = (unsigned)(ga & 0xFFFFFFFFu);         // global_addr[31:0]
    g0[3] = (unsigned)((ga >> 32) & 0x1FFFFFFu)   // global_addr[56:32]
          | (2u << 30);                           // type=2 ("image")
    i32x8 g1;
    g1[0] = (int)((1u << 16)                      // data_size = 2 bytes
                | (1u << 20)                      // pad_enable
                | (padInterval << 22)
                | (padAmount << 25));
    g1[1] = (int)((d0 & 0xFFFFu) << 16);                          // tensor_dim0 lo
    g1[2] = (int)(((d0 >> 16) & 0xFFFFu) | ((d1 & 0xFFFFu) << 16)); // dim0 hi | dim1 lo
    g1[3] = (int)(((d1 >> 16) & 0xFFFFu) | ((d0 & 0xFFFFu) << 16)); // dim1 hi | tile_dim0
    g1[4] = (int)(d1 & 0xFFFFu);                                  // tile_dim1 (tile_dim2=0)
    g1[5] = (int)strideElems;                                     // tensor_dim0_stride lo32
    g1[6] = 0;
    g1[7] = 0;
    i32x4 z4 = {0, 0, 0, 0};
    i32x8 z8 = {0, 0, 0, 0, 0, 0, 0, 0};
    __builtin_amdgcn_tensor_load_to_lds(g0, g1, z4, z4, z8, 0);
}

__device__ inline void tdm_wait0() {
#if __has_builtin(__builtin_amdgcn_s_wait_tensorcnt)
    __builtin_amdgcn_s_wait_tensorcnt(0);
#else
    asm volatile("s_wait_tensorcnt 0x0" ::: "memory");
#endif
}
__device__ inline void tdm_wait2() {
#if __has_builtin(__builtin_amdgcn_s_wait_tensorcnt)
    __builtin_amdgcn_s_wait_tensorcnt(2);
#else
    asm volatile("s_wait_tensorcnt 0x2" ::: "memory");
#endif
}
#endif

// ---------------- kernel 0a: fp32 -> bf16 (same layout) ----------------
__global__ void cvt_bf16_kernel(const float* __restrict__ in,
                                __bf16* __restrict__ out, int n) {
    int i = blockIdx.x * blockDim.x + threadIdx.x;
    if (i < n) out[i] = (__bf16)in[i];
}

// ---------------- kernel 0b: fp32 -> bf16 with per-matrix transpose -------
// in: [nMat][R][N] -> out: [nMat][N][R]   (output-linear indexing)
__global__ void cvt_transpose_kernel(const float* __restrict__ in,
                                     __bf16* __restrict__ out,
                                     int R, int N, int total) {
    int i = blockIdx.x * blockDim.x + threadIdx.x;
    if (i >= total) return;
    int mat = i / (R * N);
    int rem = i - mat * (R * N);
    int n = rem / R;
    int k = rem - n * R;
    out[i] = (__bf16)in[(size_t)mat * R * N + (size_t)k * N + n];
}

// ---------------- kernel 1: QKV projection GEMM ----------------
// Per (b, head): out[T x HS] = X[T x C] @ Wt[HS x C]^T  (Q prescaled by 1/32)
// Weights pre-transposed (n-major). V written transposed [HS x T].
// grid: (T/256, H, B*3) ; block: 256 (8 waves, 32x64 output tile each)
__global__ __launch_bounds__(256)
void qkv_gemm_kernel(const __bf16* __restrict__ xb,
                     const __bf16* __restrict__ wqt,
                     const __bf16* __restrict__ wkt,
                     const __bf16* __restrict__ wvt,
                     __bf16* __restrict__ Qb,
                     __bf16* __restrict__ Kb,
                     __bf16* __restrict__ Vtb) {
    constexpr int LDA = 40;   // 32 + 8 pad halves (TDM pad: 16DW stored -> +4DW)
    constexpr int LDB = 40;
    __shared__ __bf16 ldsA[2][256 * LDA];
    __shared__ __bf16 ldsB[2][64 * LDB];

    const int tid   = threadIdx.x;
    const int lane  = tid & 31;
    const int wv_id = tid >> 5;
    const int tBase = blockIdx.x * 256;
    const int h     = blockIdx.y;
    const int b     = blockIdx.z / 3;
    const int which = blockIdx.z % 3;

    const __bf16* Wt = (which == 0 ? wqt : which == 1 ? wkt : wvt) + (size_t)h * HS * C;
    const float oscale = (which == 0) ? 0.03125f : 1.0f;   // C^-0.5 = 1/32

    const __bf16* X = xb + (size_t)b * T * C + (size_t)tBase * C;

    v8f acc[2][4];
    for (int s = 0; s < 2; ++s)
        for (int nt = 0; nt < 4; ++nt) acc[s][nt] = zero8();

    const int mBase = wv_id * 32;
    constexpr int STEPS = C / 32;

#ifdef USE_TDM
    if (wv_id == 0) {
        tdm_load_2d(lds_off(&ldsA[0][0]), &X[0],  32, 256, C, 3u, 3u);
        tdm_load_2d(lds_off(&ldsB[0][0]), &Wt[0], 32, 64,  C, 3u, 3u);
    }
#endif
#pragma unroll 1
    for (int i = 0; i < STEPS; ++i) {
        const int cur = i & 1;
        const __bf16* curA = cur ? &ldsA[1][0] : &ldsA[0][0];
        const __bf16* curB = cur ? &ldsB[1][0] : &ldsB[0][0];
#ifdef USE_TDM
        if (wv_id == 0) {
            if (i + 1 < STEPS) {
                const int k1 = (i + 1) * 32;
                tdm_load_2d(lds_off(&ldsA[cur ^ 1][0]), &X[k1],  32, 256, C, 3u, 3u);
                tdm_load_2d(lds_off(&ldsB[cur ^ 1][0]), &Wt[k1], 32, 64,  C, 3u, 3u);
                tdm_wait2();        // in-order: waits for the 2 current tiles
            } else {
                tdm_wait0();
            }
        }
#else
        {
            const int k0 = i * 32;
            for (int e = tid; e < 1024; e += 256) {
                int row = e >> 2, c8 = (e & 3) << 3;
                *reinterpret_cast<uint4*>(const_cast<__bf16*>(&curA[row * LDA + c8])) =
                    *reinterpret_cast<const uint4*>(&X[row * C + k0 + c8]);
            }
            for (int e = tid; e < 256; e += 256) {
                int row = e >> 2, c8 = (e & 3) << 3;
                *reinterpret_cast<uint4*>(const_cast<__bf16*>(&curB[row * LDB + c8])) =
                    *reinterpret_cast<const uint4*>(&Wt[row * C + k0 + c8]);
            }
        }
#endif
        __syncthreads();

        v16bf a0 = load_fragA(&curA[mBase * LDA], LDA, lane);
        v16bf a1 = load_fragA(&curA[(mBase + 16) * LDA], LDA, lane);
        for (int nt = 0; nt < 4; ++nt) {
            v16bf bf = load_fragB(&curB[nt * 16 * LDB], LDB, lane);
            acc[0][nt] = wmma_bf16(a0, bf, acc[0][nt]);
            acc[1][nt] = wmma_bf16(a1, bf, acc[1][nt]);
        }
        __syncthreads();
    }

    const int n    = lane & 15;
    const int mAdd = (lane >> 4) << 3;
    if (which != 2) {
        __bf16* O = (which == 0 ? Qb : Kb) + (size_t)(b * H + h) * T * HS;
        for (int s = 0; s < 2; ++s)
            for (int nt = 0; nt < 4; ++nt)
                for (int r = 0; r < 8; ++r) {
                    int row = tBase + mBase + s * 16 + r + mAdd;
                    O[(size_t)row * HS + nt * 16 + n] = (__bf16)(acc[s][nt][r] * oscale);
                }
    } else {
        // V stored transposed: Vt[d][t]
        __bf16* Ot = Vtb + (size_t)(b * H + h) * HS * T;
        for (int s = 0; s < 2; ++s)
            for (int nt = 0; nt < 4; ++nt)
                for (int r = 0; r < 8; ++r) {
                    int row = tBase + mBase + s * 16 + r + mAdd;
                    Ot[(size_t)(nt * 16 + n) * T + row] = (__bf16)acc[s][nt][r];
                }
    }
}

// ---------------- kernel 2: causal flash attention ----------------
// grid: (T/128, H, B) ; block: 256 (8 waves, 16 query rows each)
__global__ __launch_bounds__(256)
void attn_kernel(const __bf16* __restrict__ Qb,
                 const __bf16* __restrict__ Kb,
                 const __bf16* __restrict__ Vtb,
                 __bf16* __restrict__ attnb) {
    constexpr int LDK = 72;   // 64 + 8 pad (TDM: 32DW stored -> +4DW)
    constexpr int LDV = 40;   // 32 + 8 pad (TDM: 16DW stored -> +4DW)
    constexpr int LDP = 40;
    __shared__ __bf16 ldsK[2][32 * LDK];
    __shared__ __bf16 ldsVt[2][64 * LDV];
    __shared__ __bf16 ldsP[8][16 * LDP];

    const int tid   = threadIdx.x;
    const int lane  = tid & 31;
    const int wv_id = tid >> 5;
    const int qBase = blockIdx.x * 128;
    const int h     = blockIdx.y;
    const int b     = blockIdx.z;

    const __bf16* Qg  = Qb  + (size_t)(b * H + h) * T * HS;
    const __bf16* Kg  = Kb  + (size_t)(b * H + h) * T * HS;
    const __bf16* Vtg = Vtb + (size_t)(b * H + h) * HS * T;   // [HS][T]

    const __bf16* qrow = Qg + (size_t)(qBase + wv_id * 16) * HS;
    v16bf qf[2];
    qf[0] = load_fragA(qrow,      HS, lane);
    qf[1] = load_fragA(qrow + 32, HS, lane);

    v8f o[4];
    for (int nt = 0; nt < 4; ++nt) o[nt] = zero8();
    float mi[8], li[8];
    for (int r = 0; r < 8; ++r) { mi[r] = -__builtin_inff(); li[r] = 0.0f; }

    const int n     = lane & 15;
    const int mAdd  = (lane >> 4) << 3;
    const int qRow0 = qBase + wv_id * 16;

    const int numKB = (qBase + 128) >> 5;   // causal: key blocks of 32

#ifdef USE_TDM
    if (wv_id == 0) {
        tdm_load_2d(lds_off(&ldsK[0][0]),  &Kg[0],  64, 32, HS, 4u, 3u);
        tdm_load_2d(lds_off(&ldsVt[0][0]), &Vtg[0], 32, 64, T,  3u, 3u);
    }
#endif
#pragma unroll 1
    for (int kb = 0; kb < numKB; ++kb) {
        const int cur   = kb & 1;
        const int sBase = kb << 5;
        const __bf16* curK = cur ? &ldsK[1][0] : &ldsK[0][0];
        const __bf16* curV = cur ? &ldsVt[1][0] : &ldsVt[0][0];
#ifdef USE_TDM
        if (wv_id == 0) {
            if (kb + 1 < numKB) {
                const int sn = sBase + 32;
                tdm_load_2d(lds_off(&ldsK[cur ^ 1][0]),  &Kg[(size_t)sn * HS], 64, 32, HS, 4u, 3u);
                tdm_load_2d(lds_off(&ldsVt[cur ^ 1][0]), &Vtg[sn],             32, 64, T,  3u, 3u);
                tdm_wait2();
            } else {
                tdm_wait0();
            }
        }
#else
        {
            int row = tid >> 3, c8 = (tid & 7) << 3;
            *reinterpret_cast<uint4*>(const_cast<__bf16*>(&curK[row * LDK + c8])) =
                *reinterpret_cast<const uint4*>(&Kg[(size_t)(sBase + row) * HS + c8]);
            for (int e = tid; e < 512; e += 256) {
                int d = e >> 2, c = (e & 3) << 3;
                *reinterpret_cast<uint4*>(const_cast<__bf16*>(&curV[d * LDV + c])) =
                    *reinterpret_cast<const uint4*>(&Vtg[(size_t)d * T + sBase + c]);
            }
        }
#endif
        __syncthreads();

        // S = Q @ K^T  (two 16x16 tiles over 32 keys)
        v8f s0 = zero8(), s1 = zero8();
        for (int kk = 0; kk < 2; ++kk) {
            v16bf bk0 = load_fragB(&curK[0 * LDK + kk * 32], LDK, lane);
            v16bf bk1 = load_fragB(&curK[16 * LDK + kk * 32], LDK, lane);
            s0 = wmma_bf16(qf[kk], bk0, s0);
            s1 = wmma_bf16(qf[kk], bk1, s1);
        }

        // causal mask + online softmax (row stats via 16-lane shuffles)
        __bf16* Pw = ldsP[wv_id];
        for (int r = 0; r < 8; ++r) {
            int q  = qRow0 + r + mAdd;
            int c0 = sBase + n, c1 = sBase + 16 + n;
            float v0 = (c0 > q) ? -__builtin_inff() : s0[r];
            float v1 = (c1 > q) ? -__builtin_inff() : s1[r];
            float rmax = rowmax16(fmaxf(v0, v1));
            float mnew = fmaxf(mi[r], rmax);
            float alpha = (mi[r] == -__builtin_inff()) ? 0.0f : __expf(mi[r] - mnew);
            float p0 = __expf(v0 - mnew);
            float p1 = __expf(v1 - mnew);
            li[r] = li[r] * alpha + rowsum16(p0 + p1);
            mi[r] = mnew;
            for (int nt = 0; nt < 4; ++nt) o[nt][r] *= alpha;
            int m = r + mAdd;
            Pw[m * LDP + n]      = (__bf16)p0;
            Pw[m * LDP + 16 + n] = (__bf16)p1;
        }
        asm volatile("s_wait_dscnt 0x0" ::: "memory");

        // O += P @ V
        v16bf pf = load_fragA(Pw, LDP, lane);
        for (int nt = 0; nt < 4; ++nt) {
            v16bf bvf = load_fragB(&curV[nt * 16 * LDV], LDV, lane);
            o[nt] = wmma_bf16(pf, bvf, o[nt]);
        }
        __syncthreads();
    }

    for (int r = 0; r < 8; ++r) {
        float inv = 1.0f / li[r];
        int t = qRow0 + r + mAdd;
        size_t rowOff = ((size_t)b * T + t) * C + h * HS;
        for (int nt = 0; nt < 4; ++nt)
            attnb[rowOff + nt * 16 + n] = (__bf16)(o[nt][r] * inv);
    }
}

// ---------------- kernel 3: output projection + bias ----------------
// out[BT x C] = attn[BT x C] @ Wproj + bproj ; Wproj pre-transposed (n-major)
// grid: (BT/256, C/64) ; block: 256
__global__ __launch_bounds__(256)
void proj_gemm_kernel(const __bf16* __restrict__ attnb,
                      const __bf16* __restrict__ wpt,
                      const float* __restrict__ bias,
                      float* __restrict__ out) {
    constexpr int LDA = 40;
    constexpr int LDB = 40;
    __shared__ __bf16 ldsA[2][256 * LDA];
    __shared__ __bf16 ldsB[2][64 * LDB];

    const int tid   = threadIdx.x;
    const int lane  = tid & 31;
    const int wv_id = tid >> 5;
    const int rBase = blockIdx.x * 256;
    const int nBase = blockIdx.y * 64;

    const __bf16* X  = attnb + (size_t)rBase * C;
    const __bf16* Wt = wpt + (size_t)nBase * C;   // rows n, stride C

    v8f acc[2][4];
    for (int s = 0; s < 2; ++s)
        for (int nt = 0; nt < 4; ++nt) acc[s][nt] = zero8();
    const int mBase = wv_id * 32;
    constexpr int STEPS = C / 32;

#ifdef USE_TDM
    if (wv_id == 0) {
        tdm_load_2d(lds_off(&ldsA[0][0]), &X[0],  32, 256, C, 3u, 3u);
        tdm_load_2d(lds_off(&ldsB[0][0]), &Wt[0], 32, 64,  C, 3u, 3u);
    }
#endif
#pragma unroll 1
    for (int i = 0; i < STEPS; ++i) {
        const int cur = i & 1;
        const __bf16* curA = cur ? &ldsA[1][0] : &ldsA[0][0];
        const __bf16* curB = cur ? &ldsB[1][0] : &ldsB[0][0];
#ifdef USE_TDM
        if (wv_id == 0) {
            if (i + 1 < STEPS) {
                const int k1 = (i + 1) * 32;
                tdm_load_2d(lds_off(&ldsA[cur ^ 1][0]), &X[k1],  32, 256, C, 3u, 3u);
                tdm_load_2d(lds_off(&ldsB[cur ^ 1][0]), &Wt[k1], 32, 64,  C, 3u, 3u);
                tdm_wait2();
            } else {
                tdm_wait0();
            }
        }
#else
        {
            const int k0 = i * 32;
            for (int e = tid; e < 1024; e += 256) {
                int row = e >> 2, c8 = (e & 3) << 3;
                *reinterpret_cast<uint4*>(const_cast<__bf16*>(&curA[row * LDA + c8])) =
                    *reinterpret_cast<const uint4*>(&X[row * C + k0 + c8]);
            }
            for (int e = tid; e < 256; e += 256) {
                int row = e >> 2, c8 = (e & 3) << 3;
                *reinterpret_cast<uint4*>(const_cast<__bf16*>(&curB[row * LDB + c8])) =
                    *reinterpret_cast<const uint4*>(&Wt[row * C + k0 + c8]);
            }
        }
#endif
        __syncthreads();

        v16bf a0 = load_fragA(&curA[mBase * LDA], LDA, lane);
        v16bf a1 = load_fragA(&curA[(mBase + 16) * LDA], LDA, lane);
        for (int nt = 0; nt < 4; ++nt) {
            v16bf bf = load_fragB(&curB[nt * 16 * LDB], LDB, lane);
            acc[0][nt] = wmma_bf16(a0, bf, acc[0][nt]);
            acc[1][nt] = wmma_bf16(a1, bf, acc[1][nt]);
        }
        __syncthreads();
    }

    const int n    = lane & 15;
    const int mAdd = (lane >> 4) << 3;
    for (int s = 0; s < 2; ++s)
        for (int nt = 0; nt < 4; ++nt) {
            int col = nBase + nt * 16 + n;
            float bv = bias[col];
            for (int r = 0; r < 8; ++r) {
                int row = rBase + mBase + s * 16 + r + mAdd;
                out[(size_t)row * C + col] = acc[s][nt][r] + bv;
            }
        }
}

// ---------------- host launcher ----------------
extern "C" void kernel_launch(void* const* d_in, const int* in_sizes, int n_in,
                              void* d_out, int out_size, void* d_ws, size_t ws_size,
                              hipStream_t stream) {
    const float* x     = (const float*)d_in[0];
    const float* Wq    = (const float*)d_in[1];
    const float* Wk    = (const float*)d_in[2];
    const float* Wv    = (const float*)d_in[3];
    const float* Wproj = (const float*)d_in[4];
    const float* bproj = (const float*)d_in[5];
    float* out = (float*)d_out;

    char* ws = (char*)d_ws;
    const size_t nX   = (size_t)Bz * T * C;
    const size_t nW   = (size_t)H * C * HS;
    const size_t nWP  = (size_t)C * C;
    const size_t nQKV = (size_t)Bz * H * T * HS;

    __bf16* xb    = (__bf16*)ws; ws += nX   * 2;
    __bf16* wqt   = (__bf16*)ws; ws += nW   * 2;
    __bf16* wkt   = (__bf16*)ws; ws += nW   * 2;
    __bf16* wvt   = (__bf16*)ws; ws += nW   * 2;
    __bf16* wpt   = (__bf16*)ws; ws += nWP  * 2;
    __bf16* Qb    = (__bf16*)ws; ws += nQKV * 2;
    __bf16* Kb    = (__bf16*)ws; ws += nQKV * 2;
    __bf16* Vtb   = (__bf16*)ws; ws += nQKV * 2;
    __bf16* attnb = (__bf16*)ws; ws += nX   * 2;

    cvt_bf16_kernel<<<(int)((nX + 255) / 256), 256, 0, stream>>>(x, xb, (int)nX);

    // per-head weight transpose: [H][C][HS] -> [H][HS][C]
    cvt_transpose_kernel<<<(int)((nW + 255) / 256), 256, 0, stream>>>(Wq, wqt, C, HS, (int)nW);
    cvt_transpose_kernel<<<(int)((nW + 255) / 256), 256, 0, stream>>>(Wk, wkt, C, HS, (int)nW);
    cvt_transpose_kernel<<<(int)((nW + 255) / 256), 256, 0, stream>>>(Wv, wvt, C, HS, (int)nW);
    // Wproj: [C][C] -> n-major [C][C]
    cvt_transpose_kernel<<<(int)((nWP + 255) / 256), 256, 0, stream>>>(Wproj, wpt, C, C, (int)nWP);

    dim3 g1(T / 256, H, Bz * 3);
    qkv_gemm_kernel<<<g1, 256, 0, stream>>>(xb, wqt, wkt, wvt, Qb, Kb, Vtb);

    dim3 g2(T / 128, H, Bz);
    attn_kernel<<<g2, 256, 0, stream>>>(Qb, Kb, Vtb, attnb);

    dim3 g3((Bz * T) / 256, C / 64);
    proj_gemm_kernel<<<g3, 256, 0, stream>>>(attnb, wpt, bproj, out);
}
